// FPNClassifier_45732811768004
// MI455X (gfx1250) — matrix operand order repaired and checked by reading, not compiled, not run on hardware
//
#include <hip/hip_runtime.h>
#include <math.h>

typedef __attribute__((ext_vector_type(16))) __bf16 v16bf;
typedef __attribute__((ext_vector_type(8)))  __bf16 v8bf;
typedef __attribute__((ext_vector_type(8)))  float  v8f;
typedef unsigned int u32x4 __attribute__((ext_vector_type(4)));
typedef int          i32x4 __attribute__((ext_vector_type(4)));
typedef int          i32x8 __attribute__((ext_vector_type(8)));

#define BATCH  2
#define NROI   1000
#define MROWS  (BATCH * NROI)     // 2000
#define MPAD   2048
#define CCH    256
#define POOLP  7
#define K1     (CCH * POOLP * POOLP)  // 12544
#define FDIM   1024
#define NCLS   81
#define NBOX   324
#define NHEAD  448                // 81 + 324 padded to multiple of 64
#define BNEPS  1e-5f

// LDS tile geometry: rows of 32 bf16 (64B) padded by 16B -> stride 80B = 40 elems.
// 20-bank row stride => conflict-free ds_read_b128 across 16 lanes.
#define LDA    40
#define ATILE_ELEMS (128 * LDA)
#define BTILE_ELEMS (64 * LDA)

// ---------------------------------------------------------------------------
// ROI-align: pooled[m][c*49 + p] in bf16, K-major for the conv1 GEMM.
// grid = (49, 2000), block = 256 (one thread per channel)
// ---------------------------------------------------------------------------
__global__ void roipool_kernel(const float* __restrict__ rois,
                               const float* __restrict__ p2, const float* __restrict__ p3,
                               const float* __restrict__ p4, const float* __restrict__ p5,
                               __bf16* __restrict__ pooled) {
  const int p  = blockIdx.x;            // 0..48
  const int m  = blockIdx.y;            // 0..1999
  const int c  = threadIdx.x;           // 0..255
  const int b  = m / NROI;
  const int ph = p / POOLP, pw = p % POOLP;

  const float* r = rois + (size_t)m * 4;
  const float y1 = r[0], x1 = r[1], y2 = r[2], x2 = r[3];
  const float h = y2 - y1, w = x2 - x1;
  const float area = fmaxf(h * w, 1e-12f);
  int lvl = 4 + (int)rintf(log2f(sqrtf(area) / 0.21875f));   // 224/1024
  lvl = min(5, max(2, lvl));
  const int li = lvl - 2;
  const int H = 256 >> li;                                   // square maps
  const float* fmap = (li == 0) ? p2 : (li == 1) ? p3 : (li == 2) ? p4 : p5;
  const float* f = fmap + ((size_t)b * CCH + c) * (size_t)H * H;

  const float ty = ph * (1.0f / 6.0f), tx = pw * (1.0f / 6.0f);
  const float ys = (y1 + h * ty) * (float)(H - 1);
  const float xs = (x1 + w * tx) * (float)(H - 1);
  const float y0 = floorf(ys), x0 = floorf(xs);
  const float wy = ys - y0, wx = xs - x0;
  const int yi0 = min(H - 1, max(0, (int)y0));
  const int yi1 = min(H - 1, max(0, (int)y0 + 1));
  const int xi0 = min(H - 1, max(0, (int)x0));
  const int xi1 = min(H - 1, max(0, (int)x0 + 1));
  const float v00 = f[yi0 * H + xi0], v01 = f[yi0 * H + xi1];
  const float v10 = f[yi1 * H + xi0], v11 = f[yi1 * H + xi1];
  const float val = v00 * (1.f - wy) * (1.f - wx) + v01 * (1.f - wy) * wx
                  + v10 * wy * (1.f - wx)         + v11 * wy * wx;
  pooled[(size_t)m * K1 + (size_t)c * 49 + p] = (__bf16)val;
}

// ---------------------------------------------------------------------------
// Small utility kernels
// ---------------------------------------------------------------------------
__global__ void cast_bf16_kernel(const float* __restrict__ src, __bf16* __restrict__ dst, int n) {
  const int i = blockIdx.x * blockDim.x + threadIdx.x;
  if (i < n) dst[i] = (__bf16)src[i];
}

__global__ void zero_bf16_kernel(__bf16* __restrict__ dst, int n) {
  const int i = blockIdx.x * blockDim.x + threadIdx.x;
  if (i < n) dst[i] = (__bf16)0.0f;
}

// Pack logits_w (81x1024) ++ bbox_w (324x1024) ++ zero pad into headw[448][1024] bf16
__global__ void build_head_kernel(const float* __restrict__ lw, const float* __restrict__ lb,
                                  const float* __restrict__ bw, const float* __restrict__ bb,
                                  __bf16* __restrict__ hw, float* __restrict__ hb) {
  const int r = blockIdx.x;             // 0..447
  for (int k = threadIdx.x; k < FDIM; k += blockDim.x) {
    float v = 0.f;
    if (r < NCLS)             v = lw[(size_t)r * FDIM + k];
    else if (r < NCLS + NBOX) v = bw[(size_t)(r - NCLS) * FDIM + k];
    hw[(size_t)r * FDIM + k] = (__bf16)v;
  }
  if (threadIdx.x == 0) {
    float v = 0.f;
    if (r < NCLS)             v = lb[r];
    else if (r < NCLS + NBOX) v = bb[r - NCLS];
    hb[r] = v;
  }
}

// ---------------------------------------------------------------------------
// Tensor Data Mover: DMA one (tile_rows x 32) bf16 tile, row stride K elems,
// into LDS at lds_byte_off with 16B padding per 64B row (LDS stride 80B).
// D# layout per CDNA5 ISA ch.8 (groups 0/1; 2D tensor so groups 2/3 zero).
// ---------------------------------------------------------------------------
__device__ __forceinline__ unsigned lds_off_u32(const void* p) {
  // generic pointer to LDS: addr[31:0] is the LDS byte offset
  return (unsigned)(unsigned long long)p;
}

__device__ __forceinline__ void tdm_load_2d(const __bf16* gsrc, unsigned lds_byte_off,
                                            int kstride, int tile_rows, int tensor_rows) {
  const unsigned long long ga = (unsigned long long)(const void*)gsrc;
  u32x4 g0;
  g0[0] = 1u;                                            // count=1, user descriptor
  g0[1] = lds_byte_off;                                  // lds_addr
  g0[2] = (unsigned)ga;                                  // global_addr[31:0]
  g0[3] = (unsigned)((ga >> 32) & 0x01ffffffu) | (2u << 30);  // addr[56:32] | type=2
  i32x8 g1;
  // data_size=1 (2B) | pad_enable | pad_interval=3 (64B) | pad_amount=3 (16B)
  g1[0] = (int)((1u << 16) | (1u << 20) | (3u << 22) | (3u << 25));
  g1[1] = (int)(((unsigned)kstride & 0xffffu) << 16);    // tensor_dim0[15:0]
  g1[2] = (int)((((unsigned)kstride >> 16) & 0xffffu) |
                (((unsigned)tensor_rows & 0xffffu) << 16)); // dim0[31:16] | dim1[15:0]
  g1[3] = (int)(32u << 16);                              // dim1[31:16]=0 | tile_dim0=32
  g1[4] = (int)((unsigned)tile_rows & 0xffffu);          // tile_dim1 | tile_dim2=0
  g1[5] = kstride;                                       // tensor_dim0_stride[31:0]
  g1[6] = 0;                                             // stride hi | dim1_stride lo
  g1[7] = 0;
  const i32x4 gz = {0, 0, 0, 0};
#if defined(__clang_major__) && (__clang_major__ >= 23)
  const i32x8 gz8 = {0, 0, 0, 0, 0, 0, 0, 0};
  __builtin_amdgcn_tensor_load_to_lds(g0, g1, gz, gz, gz8, 0);
#else
  __builtin_amdgcn_tensor_load_to_lds(g0, g1, gz, gz, 0);
#endif
}

// ---------------------------------------------------------------------------
// bf16 WMMA GEMM with TDM-staged LDS double buffering:
//   Out[M][N] = A[M][K] * Bw[N][K]^T + bias[N]
// block = 256 threads (8 waves); wave tile 32x32; block tile 128x64.
// grid = (M/128, N/64); K multiple of 32.
// ---------------------------------------------------------------------------
__device__ __forceinline__ v16bf load_frag_a(const __bf16* p) {
  // lanes 0-15: K = kb+0..7 and kb+16..23 ; lanes 16-31: kb = +8
  v8bf lo = *reinterpret_cast<const v8bf*>(p);
  v8bf hi = *reinterpret_cast<const v8bf*>(p + 16);
  v16bf r;
#pragma unroll
  for (int i = 0; i < 8; ++i) { r[i] = lo[i]; r[i + 8] = hi[i]; }
  return r;
}

__device__ __forceinline__ v16bf load_frag_b(const __bf16* p) {
  // lanes 0-15: K = kb+0..7 and kb+16..23 ; lanes 16-31: kb = +8
  v8bf lo = *reinterpret_cast<const v8bf*>(p);
  v8bf hi = *reinterpret_cast<const v8bf*>(p + 16);
  v16bf r;
#pragma unroll
  for (int i = 0; i < 8; ++i) { r[i] = lo[i]; r[i + 8] = hi[i]; }
  return r;
}

__global__ void __launch_bounds__(256)
gemm_bf16_tdm_kernel(const __bf16* __restrict__ A, const __bf16* __restrict__ Bw,
                     const float* __restrict__ bias, float* __restrict__ Out,
                     int K, int Nn, int Mtot) {
  __shared__ __align__(128) __bf16 smem[2][ATILE_ELEMS + BTILE_ELEMS];

  const int lane = threadIdx.x & 31;
  const int wave = threadIdx.x >> 5;
  const int wm = wave & 3, wn = wave >> 2;
  const int l16 = lane & 15;
  const int lhi = lane >> 4;
  const int row0 = blockIdx.x * 128 + wm * 32;   // global row of this wave's tile
  const int col0 = blockIdx.y * 64 + wn * 32;    // global col of this wave's tile

  const __bf16* Ag = A  + (size_t)(blockIdx.x * 128) * K;
  const __bf16* Bg = Bw + (size_t)(blockIdx.y * 64) * K;

  const v8f zf = {0.f, 0.f, 0.f, 0.f, 0.f, 0.f, 0.f, 0.f};
  v8f acc[2][2];
#pragma unroll
  for (int i = 0; i < 2; ++i)
#pragma unroll
    for (int j = 0; j < 2; ++j) acc[i][j] = zf;

  // in-tile LDS fragment bases (elements)
  const int aOff0 = (wm * 32 + l16) * LDA + (lhi << 3);
  const int aOff1 = (wm * 32 + 16 + l16) * LDA + (lhi << 3);
  const int bOff0 = ATILE_ELEMS + (wn * 32 + l16) * LDA + (lhi << 3);
  const int bOff1 = ATILE_ELEMS + (wn * 32 + 16 + l16) * LDA + (lhi << 3);

  // prologue: TDM-stage k0=0 tiles into buffer 0
  if (wave == 0) {
    tdm_load_2d(Ag, lds_off_u32(&smem[0][0]),           K, 128, Mtot);
    tdm_load_2d(Bg, lds_off_u32(&smem[0][ATILE_ELEMS]), K, 64,  Nn);
    __builtin_amdgcn_s_wait_tensorcnt(0);
  }
  __syncthreads();

  int buf = 0;
  for (int k0 = 0; k0 < K; k0 += 32) {
    const int nb = buf ^ 1;
    const bool more = (k0 + 32) < K;
    if (wave == 0 && more) {   // prefetch next K-slab via TDM while we compute
      tdm_load_2d(Ag + (k0 + 32), lds_off_u32(&smem[nb][0]),           K, 128, Mtot);
      tdm_load_2d(Bg + (k0 + 32), lds_off_u32(&smem[nb][ATILE_ELEMS]), K, 64,  Nn);
    }

    const __bf16* sb = &smem[buf][0];
    v16bf a0 = load_frag_a(sb + aOff0);
    v16bf a1 = load_frag_a(sb + aOff1);
    v16bf b0 = load_frag_b(sb + bOff0);
    v16bf b1 = load_frag_b(sb + bOff1);
    acc[0][0] = __builtin_amdgcn_wmma_f32_16x16x32_bf16(false, a0, false, b0, (short)0, acc[0][0], false, false);
    acc[0][1] = __builtin_amdgcn_wmma_f32_16x16x32_bf16(false, a0, false, b1, (short)0, acc[0][1], false, false);
    acc[1][0] = __builtin_amdgcn_wmma_f32_16x16x32_bf16(false, a1, false, b0, (short)0, acc[1][0], false, false);
    acc[1][1] = __builtin_amdgcn_wmma_f32_16x16x32_bf16(false, a1, false, b1, (short)0, acc[1][1], false, false);

    if (wave == 0 && more) __builtin_amdgcn_s_wait_tensorcnt(0);
    __syncthreads();
    buf = nb;
  }

  // C/D layout: lane n = col + (lane&15); VGPR r -> m = row + (lane>=16 ? 8 : 0) + r
#pragma unroll
  for (int mi = 0; mi < 2; ++mi) {
    const int mb = row0 + mi * 16 + (lhi << 3) - wm * 32 + wm * 32;  // = row0 tile + ...
#pragma unroll
    for (int ni = 0; ni < 2; ++ni) {
      const int n = col0 + ni * 16 + l16;
      const float bia = bias[n];
#pragma unroll
      for (int r8 = 0; r8 < 8; ++r8)
        Out[(size_t)(row0 + mi * 16 + (lhi << 3) + r8) * Nn + n] = acc[mi][ni][r8] + bia;
    }
    (void)mb;
  }
}

// ---------------------------------------------------------------------------
// BatchNorm over the ROI axis: mean/rstd per (b, f) over the 1000 real rows.
// grid = (FDIM, BATCH), block = 128
// ---------------------------------------------------------------------------
__global__ void bnstats_kernel(const float* __restrict__ X,
                               float* __restrict__ meanO, float* __restrict__ rstdO) {
  const int f = blockIdx.x;
  const int b = blockIdx.y;
  __shared__ float ss[128], sq[128];
  float s = 0.f, q = 0.f;
  for (int n = threadIdx.x; n < NROI; n += 128) {
    const float v = X[(size_t)(b * NROI + n) * FDIM + f];
    s += v; q += v * v;
  }
  ss[threadIdx.x] = s; sq[threadIdx.x] = q;
  __syncthreads();
  for (int st = 64; st > 0; st >>= 1) {
    if (threadIdx.x < st) { ss[threadIdx.x] += ss[threadIdx.x + st]; sq[threadIdx.x] += sq[threadIdx.x + st]; }
    __syncthreads();
  }
  if (threadIdx.x == 0) {
    const float mu  = ss[0] * (1.0f / NROI);
    const float var = sq[0] * (1.0f / NROI) - mu * mu;
    meanO[b * FDIM + f] = mu;
    rstdO[b * FDIM + f] = rsqrtf(var + BNEPS);
  }
}

// BN+ReLU then cast to bf16 (zero for pad rows). grid = MPAD*FDIM/256
__global__ void bnrelu_kernel(const float* __restrict__ X, const float* __restrict__ meanI,
                              const float* __restrict__ rstdI, const float* __restrict__ g,
                              const float* __restrict__ bt, __bf16* __restrict__ Out) {
  const int i = blockIdx.x * blockDim.x + threadIdx.x;
  const int m = i >> 10;       // /FDIM
  const int f = i & 1023;
  float o = 0.f;
  if (m < MROWS) {
    const int b = (m >= NROI) ? 1 : 0;
    const float v = X[i];
    o = fmaxf(0.f, g[f] * (v - meanI[b * FDIM + f]) * rstdI[b * FDIM + f] + bt[f]);
  }
  Out[i] = (__bf16)o;
}

// ---------------------------------------------------------------------------
// Softmax + scatter into d_out: [logits | probs | bbox]
// grid = 2000, block = 128
// ---------------------------------------------------------------------------
__global__ void epilogue_kernel(const float* __restrict__ ho, float* __restrict__ out) {
  const int m = blockIdx.x;
  const float* row = ho + (size_t)m * NHEAD;
  __shared__ float red[128];
  const int t = threadIdx.x;
  const float v = (t < NCLS) ? row[t] : -3.0e38f;
  red[t] = v;
  __syncthreads();
  for (int st = 64; st > 0; st >>= 1) { if (t < st) red[t] = fmaxf(red[t], red[t + st]); __syncthreads(); }
  const float mx = red[0];
  __syncthreads();
  const float e = (t < NCLS) ? expf(v - mx) : 0.f;
  red[t] = e;
  __syncthreads();
  for (int st = 64; st > 0; st >>= 1) { if (t < st) red[t] += red[t + st]; __syncthreads(); }
  const float inv = 1.0f / red[0];
  if (t < NCLS) {
    out[(size_t)m * NCLS + t]          = v;            // logits
    out[162000 + (size_t)m * NCLS + t] = e * inv;      // probs
  }
  for (int j = t; j < NBOX; j += 128)
    out[324000 + (size_t)m * NBOX + j] = row[NCLS + j];  // bbox (bias already added)
}

// ---------------------------------------------------------------------------
extern "C" void kernel_launch(void* const* d_in, const int* in_sizes, int n_in,
                              void* d_out, int out_size, void* d_ws, size_t ws_size,
                              hipStream_t stream) {
  (void)in_sizes; (void)n_in; (void)out_size; (void)ws_size;
  const float* rois    = (const float*)d_in[0];
  const float* p2      = (const float*)d_in[1];
  const float* p3      = (const float*)d_in[2];
  const float* p4      = (const float*)d_in[3];
  const float* p5      = (const float*)d_in[4];
  const float* conv1_w = (const float*)d_in[5];
  const float* conv1_b = (const float*)d_in[6];
  const float* bn1_g   = (const float*)d_in[7];
  const float* bn1_b   = (const float*)d_in[8];
  const float* conv2_w = (const float*)d_in[9];
  const float* conv2_b = (const float*)d_in[10];
  const float* bn2_g   = (const float*)d_in[11];
  const float* bn2_b   = (const float*)d_in[12];
  const float* lgw     = (const float*)d_in[13];
  const float* lgb     = (const float*)d_in[14];
  const float* bxw     = (const float*)d_in[15];
  const float* bxb     = (const float*)d_in[16];
  float* out = (float*)d_out;

  char* ws = (char*)d_ws;
  size_t off = 0;
  auto take = [&](size_t bytes) -> char* {
    char* p = ws + off;
    off = (off + bytes + 255) & ~(size_t)255;
    return p;
  };
  __bf16* pooled  = (__bf16*)take((size_t)MPAD * K1 * 2);
  __bf16* w1t     = (__bf16*)take((size_t)FDIM * K1 * 2);
  float*  x1      = (float*) take((size_t)MPAD * FDIM * 4);
  float*  mean1   = (float*) take((size_t)BATCH * FDIM * 4);
  float*  rstd1   = (float*) take((size_t)BATCH * FDIM * 4);
  __bf16* h1      = (__bf16*)take((size_t)MPAD * FDIM * 2);
  __bf16* w2t     = (__bf16*)take((size_t)FDIM * FDIM * 2);
  float*  x2      = (float*) take((size_t)MPAD * FDIM * 4);
  float*  mean2   = (float*) take((size_t)BATCH * FDIM * 4);
  float*  rstd2   = (float*) take((size_t)BATCH * FDIM * 4);
  __bf16* sharedb = (__bf16*)take((size_t)MPAD * FDIM * 2);
  __bf16* headw   = (__bf16*)take((size_t)NHEAD * FDIM * 2);
  float*  headb   = (float*) take((size_t)NHEAD * 4);
  float*  headout = (float*) take((size_t)MPAD * NHEAD * 4);

  // Weight prep (bf16, K-contiguous rows already)
  {
    const int n1 = FDIM * K1;
    cast_bf16_kernel<<<(n1 + 255) / 256, 256, 0, stream>>>(conv1_w, w1t, n1);
    const int n2 = FDIM * FDIM;
    cast_bf16_kernel<<<(n2 + 255) / 256, 256, 0, stream>>>(conv2_w, w2t, n2);
    build_head_kernel<<<NHEAD, 256, 0, stream>>>(lgw, lgb, bxw, bxb, headw, headb);
  }

  // ROI align -> pooled bf16 ; zero pad rows 2000..2047
  roipool_kernel<<<dim3(49, MROWS), 256, 0, stream>>>(rois, p2, p3, p4, p5, pooled);
  {
    const int npad = (MPAD - MROWS) * K1;
    zero_bf16_kernel<<<(npad + 255) / 256, 256, 0, stream>>>(pooled + (size_t)MROWS * K1, npad);
  }

  // conv1 GEMM: (2048 x 12544) * (12544 x 1024)
  gemm_bf16_tdm_kernel<<<dim3(MPAD / 128, FDIM / 64), 256, 0, stream>>>(pooled, w1t, conv1_b, x1, K1, FDIM, MPAD);
  bnstats_kernel<<<dim3(FDIM, BATCH), 128, 0, stream>>>(x1, mean1, rstd1);
  bnrelu_kernel<<<(MPAD * FDIM) / 256, 256, 0, stream>>>(x1, mean1, rstd1, bn1_g, bn1_b, h1);

  // fc2 GEMM: (2048 x 1024) * (1024 x 1024)
  gemm_bf16_tdm_kernel<<<dim3(MPAD / 128, FDIM / 64), 256, 0, stream>>>(h1, w2t, conv2_b, x2, FDIM, FDIM, MPAD);
  bnstats_kernel<<<dim3(FDIM, BATCH), 128, 0, stream>>>(x2, mean2, rstd2);
  bnrelu_kernel<<<(MPAD * FDIM) / 256, 256, 0, stream>>>(x2, mean2, rstd2, bn2_g, bn2_b, sharedb);

  // heads GEMM: (2048 x 1024) * (1024 x 448)
  gemm_bf16_tdm_kernel<<<dim3(MPAD / 128, NHEAD / 64), 256, 0, stream>>>(sharedb, headw, headb, headout, FDIM, NHEAD, MPAD);

  // softmax + scatter
  epilogue_kernel<<<MROWS, 128, 0, stream>>>(headout, out);
}
